// Actor_80891414053633
// MI455X (gfx1250) — matrix operand order repaired and checked by reading, not compiled, not run on hardware
//
#include <hip/hip_runtime.h>

// MI455X / gfx1250, wave32. Heavy GEMMs run through v_wmma_f32_16x16x32_bf16.
// A fragments: two contiguous 16B runs per lane (native ISA A-layout, b128 loads).
// B: pre-packed fragment-major (one contiguous 32B load per lane per tile).
// wmma_gemm_seg fuses gather (A row indices), concat (128-col segments),
// fp32->bf16 conversion, and scatter-atomicAdd epilogue into the GEMM itself.
typedef __bf16 bf16_t;
typedef __attribute__((ext_vector_type(16))) __bf16 v16bf;
typedef __attribute__((ext_vector_type(8)))  __bf16 v8bf;
typedef __attribute__((ext_vector_type(8)))  float  v8f;

#define FLAG_RELU  1
#define FLAG_LOADC 2

static __device__ __forceinline__ float sigmoidf_(float x) { return 1.0f / (1.0f + expf(-x)); }

union AFrag { v16bf v; v8bf h[2]; };

template <int NTILES>
static __device__ __forceinline__ void accum_seg_bf16(
    v8f (&acc)[NTILES], const bf16_t* __restrict__ p,
    const bf16_t* __restrict__ Bbase, size_t Bkstep, int ktbase)
{
  for (int kk2 = 0; kk2 < 128; kk2 += 32) {
    AFrag a;
    a.h[0] = *(const v8bf*)(p + kk2);
    a.h[1] = *(const v8bf*)(p + kk2 + 16);
    const bf16_t* bp = Bbase + (size_t)(ktbase + (kk2 >> 5)) * Bkstep;
#pragma unroll
    for (int t = 0; t < NTILES; ++t) {
      const v16bf b = *(const v16bf*)(bp + (size_t)t * 512);
      acc[t] = __builtin_amdgcn_wmma_f32_16x16x32_bf16(false, a.v, false, b,
                                                       (short)0, acc[t], false, false);
    }
  }
}

template <int NTILES>
static __device__ __forceinline__ void accum_seg_f32(
    v8f (&acc)[NTILES], const float* __restrict__ p,
    const bf16_t* __restrict__ Bbase, size_t Bkstep, int ktbase)
{
  for (int kk2 = 0; kk2 < 128; kk2 += 32) {
    v8f f0 = *(const v8f*)(p + kk2);
    v8f f1 = *(const v8f*)(p + kk2 + 16);
    AFrag a;
#pragma unroll
    for (int q = 0; q < 8; ++q) {
      a.v[q]     = (bf16_t)f0[q];
      a.v[8 + q] = (bf16_t)f1[q];
    }
    const bf16_t* bp = Bbase + (size_t)(ktbase + (kk2 >> 5)) * Bkstep;
#pragma unroll
    for (int t = 0; t < NTILES; ++t) {
      const v16bf b = *(const v16bf*)(bp + (size_t)t * 512);
      acc[t] = __builtin_amdgcn_wmma_f32_16x16x32_bf16(false, a.v, false, b,
                                                       (short)0, acc[t], false, false);
    }
  }
}

// ---------------------------------------------------------------------------
// Plain GEMM: C[M,Nvalid] = A[M,K] @ B[K,Npad] (+bias)(+ReLU)(+=C).
// K % 32 == 0, M % 16 == 0, (Npad/16) % NTILES == 0.
// ---------------------------------------------------------------------------
template <int NTILES>
__global__ void __launch_bounds__(128) wmma_gemm_bf16(
    const bf16_t* __restrict__ A, int lda,
    const bf16_t* __restrict__ Bp, int Npad,
    const float* __restrict__ bias,
    float* __restrict__ Cf, bf16_t* __restrict__ Cbf, int ldc,
    int M, int Nvalid, int K, int flags)
{
  const int lane = threadIdx.x & 31;
  const int wave = threadIdx.x >> 5;
  const int g    = lane >> 4;
  const int hl   = lane & 15;
  const int row0 = blockIdx.y * 64 + wave * 16;
  if (row0 >= M) return;
  const int ntiles = Npad >> 4;
  const int nt0    = blockIdx.x * NTILES;

  v8f acc[NTILES];
#pragma unroll
  for (int t = 0; t < NTILES; ++t) acc[t] = {};

  if (flags & FLAG_LOADC) {
#pragma unroll
    for (int t = 0; t < NTILES; ++t) {
      const int col = (nt0 + t) * 16 + hl;
#pragma unroll
      for (int r = 0; r < 8; ++r) {
        const int row = row0 + g * 8 + r;
        acc[t][r] = (col < Nvalid) ? Cf[(size_t)row * ldc + col] : 0.0f;
      }
    }
  }

  const bf16_t* Arow   = A + (size_t)(row0 + hl) * lda + g * 8;
  const bf16_t* Bbase  = Bp + ((size_t)nt0 * 32 + lane) * 16;
  const size_t  Bkstep = (size_t)ntiles * 512;

  for (int kk = 0; kk < K; kk += 32) {
    AFrag a;
    a.h[0] = *(const v8bf*)(Arow + kk);
    a.h[1] = *(const v8bf*)(Arow + kk + 16);
    const bf16_t* bp = Bbase + (size_t)(kk >> 5) * Bkstep;
    v16bf b[NTILES];
#pragma unroll
    for (int t = 0; t < NTILES; ++t)
      b[t] = *(const v16bf*)(bp + (size_t)t * 512);
#pragma unroll
    for (int t = 0; t < NTILES; ++t)
      acc[t] = __builtin_amdgcn_wmma_f32_16x16x32_bf16(false, a.v, false, b[t],
                                                       (short)0, acc[t], false, false);
  }

#pragma unroll
  for (int t = 0; t < NTILES; ++t) {
    const int col = (nt0 + t) * 16 + hl;
    if (col < Nvalid) {
#pragma unroll
      for (int r = 0; r < 8; ++r) {
        const int row = row0 + g * 8 + r;
        float v = acc[t][r];
        if (bias) v += bias[col];
        if (flags & FLAG_RELU) v = fmaxf(v, 0.0f);
        if (Cf)  Cf[(size_t)row * ldc + col] = v;
        if (Cbf) Cbf[(size_t)row * ldc + col] = (bf16_t)v;
      }
    }
  }
}

// ---------------------------------------------------------------------------
// Segmented gather/scatter GEMM. A = [seg0 | seg1 | seg2], each 128 cols wide,
// row-stride 128, with optional per-segment row-index gather and optional fp32
// source (segf32mask bit). Epilogue: bias+ReLU then either store (Cf/Cbf) or
// atomicAdd into Cf[scat[row]*ldc + col] (fused segment_sum).
// ---------------------------------------------------------------------------
template <int NTILES>
__global__ void __launch_bounds__(128) wmma_gemm_seg(
    const bf16_t* __restrict__ A0, const int* __restrict__ i0,
    const bf16_t* __restrict__ A1, const int* __restrict__ i1,
    const bf16_t* __restrict__ A2, const int* __restrict__ i2,
    int segf32mask,
    const bf16_t* __restrict__ Bp, int Npad,
    const float* __restrict__ bias,
    float* __restrict__ Cf, bf16_t* __restrict__ Cbf, int ldc,
    const int* __restrict__ scat,
    int M, int Nvalid, int flags)
{
  const int lane = threadIdx.x & 31;
  const int wave = threadIdx.x >> 5;
  const int g    = lane >> 4;
  const int hl   = lane & 15;
  const int row0 = blockIdx.y * 64 + wave * 16;
  if (row0 >= M) return;
  const int ntiles = Npad >> 4;
  const int nt0    = blockIdx.x * NTILES;
  const int arow   = row0 + hl;

  const bf16_t* Bbase  = Bp + ((size_t)nt0 * 32 + lane) * 16;
  const size_t  Bkstep = (size_t)ntiles * 512;

  v8f acc[NTILES];
#pragma unroll
  for (int t = 0; t < NTILES; ++t) acc[t] = {};

  {  // segment 0 (always bf16)
    const size_t r = (size_t)(i0 ? i0[arow] : arow);
    accum_seg_bf16<NTILES>(acc, A0 + r * 128 + g * 8, Bbase, Bkstep, 0);
  }
  if (A1) {  // segment 1 (bf16 or fp32)
    const size_t r = (size_t)(i1 ? i1[arow] : arow);
    if (segf32mask & 2)
      accum_seg_f32<NTILES>(acc, (const float*)A1 + r * 128 + g * 8, Bbase, Bkstep, 4);
    else
      accum_seg_bf16<NTILES>(acc, A1 + r * 128 + g * 8, Bbase, Bkstep, 4);
  }
  if (A2) {  // segment 2 (bf16)
    const size_t r = (size_t)(i2 ? i2[arow] : arow);
    accum_seg_bf16<NTILES>(acc, A2 + r * 128 + g * 8, Bbase, Bkstep, 8);
  }

#pragma unroll
  for (int t = 0; t < NTILES; ++t) {
    const int col = (nt0 + t) * 16 + hl;
    if (col < Nvalid) {
#pragma unroll
      for (int r = 0; r < 8; ++r) {
        const int row = row0 + g * 8 + r;
        float v = acc[t][r];
        if (bias) v += bias[col];
        if (flags & FLAG_RELU) v = fmaxf(v, 0.0f);
        if (scat) {
          atomicAdd(&Cf[(size_t)scat[row] * ldc + col], v);
        } else {
          if (Cf)  Cf[(size_t)row * ldc + col] = v;
          if (Cbf) Cbf[(size_t)row * ldc + col] = (bf16_t)v;
        }
      }
    }
  }
}

// ---------------------------------------------------------------------------
// Pack fp32 weight [srcK, srcN] -> bf16 fragment-major [Kpad/32][Npad/16][32][16]
// ---------------------------------------------------------------------------
__global__ void pack_B(const float* __restrict__ W, bf16_t* __restrict__ dst,
                       int srcK, int srcN, int Kpad, int Npad)
{
  long idx = (long)blockIdx.x * blockDim.x + threadIdx.x;
  long total = (long)Kpad * Npad;
  if (idx >= total) return;
  const int ntiles = Npad >> 4;
  const int e  = (int)(idx & 15);
  const int l  = (int)((idx >> 4) & 31);
  const long t = idx >> 9;
  const int nt = (int)(t % ntiles);
  const int kt = (int)(t / ntiles);
  const int K   = kt * 32 + (l >> 4) * 8 + (e & 7) + (e >> 3) * 16;
  const int col = nt * 16 + (l & 15);
  float v = (K < srcK && col < srcN) ? W[(size_t)K * srcN + col] : 0.0f;
  dst[idx] = (bf16_t)v;
}

// ---------------------------------------------------------------------------
// Elementwise helpers
// ---------------------------------------------------------------------------
__global__ void cast_pad2d(const float* __restrict__ src, bf16_t* __restrict__ dst,
                           int srcR, int srcC, int dstR, int dstC)
{
  long idx = (long)blockIdx.x * blockDim.x + threadIdx.x;
  long total = (long)dstR * dstC;
  if (idx >= total) return;
  int r = (int)(idx / dstC), c = (int)(idx % dstC);
  float v = (r < srcR && c < srcC) ? src[(size_t)r * srcC + c] : 0.0f;
  dst[idx] = (bf16_t)v;
}

__global__ void fill_f32(float* __restrict__ p, float v, long n)
{
  long idx = (long)blockIdx.x * blockDim.x + threadIdx.x;
  if (idx < n) p[idx] = v;
}

__global__ void lstm_gate(const float* __restrict__ z, float* __restrict__ h,
                          float* __restrict__ c, int B)
{
  long idx = (long)blockIdx.x * blockDim.x + threadIdx.x;
  if (idx >= (long)B * 128) return;
  int b = (int)(idx >> 7), d = (int)(idx & 127);
  const float* zb = z + (size_t)b * 512;
  float i = sigmoidf_(zb[d]);
  float f = sigmoidf_(zb[128 + d]);
  float gg = tanhf(zb[256 + d]);
  float o = sigmoidf_(zb[384 + d]);
  float cn = f * c[idx] + i * gg;
  c[idx] = cn;
  h[idx] = o * tanhf(cn);
}

__global__ void attn_dot(const bf16_t* __restrict__ h, const float* __restrict__ q,
                         const int* __restrict__ gid, float* __restrict__ ei, long N)
{
  long n = (long)blockIdx.x * blockDim.x + threadIdx.x;
  if (n >= N) return;
  const bf16_t* hp = h + n * 128;
  const float* qp = q + (size_t)gid[n] * 128;
  float s = 0.0f;
  for (int d = 0; d < 128; ++d) s += (float)hp[d] * qp[d];
  ei[n] = s;
}

static __device__ __forceinline__ void atomicMaxF(float* addr, float val)
{
  unsigned int* ua = (unsigned int*)addr;
  unsigned int old = *ua;
  while (__uint_as_float(old) < val) {
    unsigned int assumed = old;
    old = atomicCAS(ua, assumed, __float_as_uint(val));
    if (old == assumed) break;
  }
}

__global__ void seg_max(const float* __restrict__ ei, const int* __restrict__ gid,
                        float* __restrict__ m, long N)
{
  long n = (long)blockIdx.x * blockDim.x + threadIdx.x;
  if (n >= N) return;
  atomicMaxF(&m[gid[n]], ei[n]);
}

__global__ void seg_expsum(const float* __restrict__ ei, const int* __restrict__ gid,
                           const float* __restrict__ m, float* __restrict__ ex,
                           float* __restrict__ s, long N)
{
  long n = (long)blockIdx.x * blockDim.x + threadIdx.x;
  if (n >= N) return;
  float e = expf(ei[n] - m[gid[n]]);
  ex[n] = e;
  atomicAdd(&s[gid[n]], e);
}

__global__ void seg_weighted(const bf16_t* __restrict__ h, const float* __restrict__ ex,
                             const float* __restrict__ s, const int* __restrict__ gid,
                             float* __restrict__ r, long N)
{
  long idx = (long)blockIdx.x * blockDim.x + threadIdx.x;
  if (idx >= N * 128) return;
  long n = idx >> 7; int d = (int)(idx & 127);
  int g = gid[n];
  float att = ex[n] / (s[g] + 1e-12f);
  atomicAdd(&r[(size_t)g * 128 + d], att * (float)h[idx]);
}

__global__ void qstar_update(float* __restrict__ qs, const float* __restrict__ h1,
                             const float* __restrict__ r, int B)
{
  long idx = (long)blockIdx.x * blockDim.x + threadIdx.x;
  if (idx >= (long)B * 128) return;
  int b = (int)(idx >> 7), d = (int)(idx & 127);
  qs[(size_t)b * 256 + d]       = h1[idx];
  qs[(size_t)b * 256 + 128 + d] = r[idx];
}

__global__ void head_dot(const bf16_t* __restrict__ A, const float* __restrict__ W,
                         const float* __restrict__ b, float* __restrict__ out,
                         long rows, int K, int OC)
{
  long idx = (long)blockIdx.x * blockDim.x + threadIdx.x;
  if (idx >= rows * OC) return;
  long r = idx / OC; int j = (int)(idx % OC);
  const bf16_t* a = A + r * (size_t)K;
  float s = b[j];
  for (int k = 0; k < K; ++k) s += (float)a[k] * W[(size_t)k * OC + j];
  out[idx] = s;
}

__global__ void edge_mlp(const float* __restrict__ xe, const float* __restrict__ W1,
                         const float* __restrict__ b1, const float* __restrict__ W2,
                         const float* __restrict__ b2, bf16_t* __restrict__ out, long E)
{
  long e = (long)blockIdx.x * blockDim.x + threadIdx.x;
  if (e >= E) return;
  float x[10], t[10];
  for (int i = 0; i < 10; ++i) x[i] = xe[e * 10 + i];
  for (int j = 0; j < 10; ++j) {
    float s = b1[j];
    for (int i = 0; i < 10; ++i) s += x[i] * W1[i * 10 + j];
    t[j] = fmaxf(s, 0.0f);
  }
  for (int d = 0; d < 128; ++d) {
    float s = b2[d];
    for (int j = 0; j < 10; ++j) s += t[j] * W2[j * 128 + d];
    out[(size_t)e * 128 + d] = (bf16_t)s;
  }
}

// ---------------------------------------------------------------------------
static inline int cdiv(long a, long b) { return (int)((a + b - 1) / b); }

extern "C" void kernel_launch(void* const* d_in, const int* in_sizes, int n_in,
                              void* d_out, int out_size, void* d_ws, size_t ws_size,
                              hipStream_t stream)
{
  (void)in_sizes; (void)n_in; (void)out_size; (void)ws_size;
  const long N = 200000, E = 400000, B = 4096;
  const int  Lnum = 3;

  const float* x_node = (const float*)d_in[0];
  const float* x_edge = (const float*)d_in[1];
  const int*   src    = (const int*)d_in[2];
  const int*   dstI   = (const int*)d_in[3];
  const int*   gid    = (const int*)d_in[4];
  const float* enc_node_W = (const float*)d_in[5];
  const float* enc_node_b = (const float*)d_in[6];
  const float* enc_edge_W = (const float*)d_in[7];
  const float* enc_edge_b = (const float*)d_in[8];
  const float* enc_msg_W  = (const float*)d_in[9];
  const float* enc_msg_b  = (const float*)d_in[10];
  const float* enc_upd_W  = (const float*)d_in[11];
  const float* enc_upd_b  = (const float*)d_in[12];
  const float* em_W1 = (const float*)d_in[13];
  const float* em_b1 = (const float*)d_in[14];
  const float* em_W2 = (const float*)d_in[15];
  const float* em_b2 = (const float*)d_in[16];
  const float* lstm_Wi0 = (const float*)d_in[17];
  const float* lstm_Wh0 = (const float*)d_in[18];
  const float* lstm_b0  = (const float*)d_in[19];
  const float* lstm_Wi1 = (const float*)d_in[20];
  const float* lstm_Wh1 = (const float*)d_in[21];
  const float* lstm_b1  = (const float*)d_in[22];
  const float* act_W1 = (const float*)d_in[23];
  const float* act_b1 = (const float*)d_in[24];
  const float* act_W2 = (const float*)d_in[25];
  const float* act_b2 = (const float*)d_in[26];
  const float* del_W1 = (const float*)d_in[27];
  const float* del_b1 = (const float*)d_in[28];
  const float* del_W2 = (const float*)d_in[29];
  const float* del_b2 = (const float*)d_in[30];
  const float* add_W1 = (const float*)d_in[31];
  const float* add_b1 = (const float*)d_in[32];
  const float* add_W2 = (const float*)d_in[33];
  const float* add_b2 = (const float*)d_in[34];
  const float* arm_W1 = (const float*)d_in[35];
  const float* arm_b1 = (const float*)d_in[36];
  const float* arm_W2 = (const float*)d_in[37];
  const float* arm_b2 = (const float*)d_in[38];

  float* out = (float*)d_out;
  float* out_act = out;                 // [B,2]
  float* out_del = out + 8192;          // [E,1]
  float* out_add = out + 408192;        // [N,1]
  float* out_arm = out + 608192;        // [N,200]

  char* ws = (char*)d_ws; size_t off = 0;
  auto alloc = [&](size_t bytes) -> void* {
    void* p = ws + off; off = (off + bytes + 255) & ~(size_t)255; return p;
  };
  bf16_t* xnode_bf = (bf16_t*)alloc(N * 64 * 2);
  bf16_t* xedge_bf = (bf16_t*)alloc(E * 32 * 2);
  bf16_t* hbf      = (bf16_t*)alloc(4 * N * 128 * 2);   // final per-encoder node states
  bf16_t* tmpH     = (bf16_t*)alloc(N * 128 * 2);       // ping-pong buffer
  bf16_t* e_bf     = (bf16_t*)alloc(E * 128 * 2);
  float*  agg      = (float*)alloc(N * 128 * 4);
  bf16_t* delH     = (bf16_t*)alloc(E * 384 * 2);
  bf16_t* nW_bf    = (bf16_t*)alloc(64 * 128 * 2);
  bf16_t* eW_bf    = (bf16_t*)alloc(32 * 128 * 2);
  bf16_t* mW_bf    = (bf16_t*)alloc(3 * 256 * 128 * 2);
  bf16_t* uW_bf    = (bf16_t*)alloc(3 * 256 * 128 * 2);
  bf16_t* Wi0_bf   = (bf16_t*)alloc(256 * 512 * 2);
  bf16_t* Wh0_bf   = (bf16_t*)alloc(128 * 512 * 2);
  bf16_t* Wi1_bf   = (bf16_t*)alloc(128 * 512 * 2);
  bf16_t* Wh1_bf   = (bf16_t*)alloc(128 * 512 * 2);
  bf16_t* actW1_bf = (bf16_t*)alloc(256 * 256 * 2);
  bf16_t* delW1_bf = (bf16_t*)alloc(384 * 384 * 2);
  bf16_t* addW1_bf = (bf16_t*)alloc(128 * 128 * 2);
  bf16_t* armW1_bf = (bf16_t*)alloc(128 * 128 * 2);
  bf16_t* armW2_bf = (bf16_t*)alloc(128 * 208 * 2);
  float*  q_star   = (float*)alloc(B * 256 * 4);
  bf16_t* qs_bf    = (bf16_t*)alloc(B * 256 * 2);
  float*  h0 = (float*)alloc(B * 128 * 4);
  float*  c0 = (float*)alloc(B * 128 * 4);
  float*  h1 = (float*)alloc(B * 128 * 4);
  float*  c1 = (float*)alloc(B * 128 * 4);
  bf16_t* hB_bf = (bf16_t*)alloc(B * 128 * 2);
  float*  zbuf  = (float*)alloc(B * 512 * 4);
  float*  ei    = (float*)alloc(N * 4);
  float*  exv   = (float*)alloc(N * 4);
  float*  mB    = (float*)alloc(B * 4);
  float*  sB    = (float*)alloc(B * 4);
  float*  rB    = (float*)alloc(B * 128 * 4);
  bf16_t* actH_bf  = (bf16_t*)alloc(B * 256 * 2);
  bf16_t* hedge_bf = (bf16_t*)alloc(E * 128 * 2);
  bf16_t* addH_bf  = (bf16_t*)alloc(N * 128 * 2);
  bf16_t* armT_bf  = (bf16_t*)alloc(N * 128 * 2);

  const int TB = 256;
  auto ew = [&](long n) { return dim3((unsigned)cdiv(n, TB)); };
  auto gemm = [&](const bf16_t* A, int lda, const bf16_t* Bw, int Npad,
                  const float* bias, float* Cf, bf16_t* Cbf, int ldc,
                  long M, int Nvalid, int K, int flags) {
    const int ntiles = Npad / 16;
    if ((ntiles & 3) == 0) {
      dim3 g((unsigned)(ntiles / 4), (unsigned)cdiv(M, 64));
      wmma_gemm_bf16<4><<<g, dim3(128), 0, stream>>>(A, lda, Bw, Npad, bias, Cf, Cbf,
                                                     ldc, (int)M, Nvalid, K, flags);
    } else {
      dim3 g((unsigned)ntiles, (unsigned)cdiv(M, 64));
      wmma_gemm_bf16<1><<<g, dim3(128), 0, stream>>>(A, lda, Bw, Npad, bias, Cf, Cbf,
                                                     ldc, (int)M, Nvalid, K, flags);
    }
  };
  auto gemm_seg = [&](const bf16_t* A0, const int* i0, const bf16_t* A1, const int* i1,
                      const bf16_t* A2, const int* i2, int f32mask,
                      const bf16_t* Bw, int Npad, const float* bias,
                      float* Cf, bf16_t* Cbf, int ldc, const int* scat,
                      long M, int Nvalid, int flags) {
    dim3 g((unsigned)(Npad / 64), (unsigned)cdiv(M, 64));
    wmma_gemm_seg<4><<<g, dim3(128), 0, stream>>>(A0, i0, A1, i1, A2, i2, f32mask,
                                                  Bw, Npad, bias, Cf, Cbf, ldc, scat,
                                                  (int)M, Nvalid, flags);
  };
  auto cast = [&](const float* s, bf16_t* d, int sr, int sc, int dr, int dc) {
    cast_pad2d<<<ew((long)dr * dc), TB, 0, stream>>>(s, d, sr, sc, dr, dc);
  };
  auto pack = [&](const float* s, bf16_t* d, int sk, int sn, int kp, int np) {
    pack_B<<<ew((long)kp * np), TB, 0, stream>>>(s, d, sk, sn, kp, np);
  };

  // ---- phase 0: cast activations, pack all weights fragment-major ----
  cast(x_node, xnode_bf, (int)N, 40, (int)N, 64);
  cast(x_edge, xedge_bf, (int)E, 10, (int)E, 32);
  pack(lstm_Wi0, Wi0_bf, 256, 512, 256, 512);
  pack(lstm_Wh0, Wh0_bf, 128, 512, 128, 512);
  pack(lstm_Wi1, Wi1_bf, 128, 512, 128, 512);
  pack(lstm_Wh1, Wh1_bf, 128, 512, 128, 512);
  pack(act_W1, actW1_bf, 256, 256, 256, 256);
  pack(del_W1, delW1_bf, 384, 384, 384, 384);
  pack(add_W1, addW1_bf, 128, 128, 128, 128);
  pack(arm_W1, armW1_bf, 128, 128, 128, 128);
  pack(arm_W2, armW2_bf, 128, 200, 128, 208);

  // ---- phase 1: four message-passing encoders (fused gather/scatter GEMMs) ----
  for (int enc = 0; enc < 4; ++enc) {
    bf16_t* h_enc = hbf + (size_t)enc * N * 128;
    pack(enc_node_W + (size_t)enc * 40 * 128, nW_bf, 40, 128, 64, 128);
    pack(enc_edge_W + (size_t)enc * 10 * 128, eW_bf, 10, 128, 32, 128);
    for (int l = 0; l < Lnum; ++l) {
      pack(enc_msg_W + ((size_t)enc * 3 + l) * 256 * 128, mW_bf + (size_t)l * 256 * 128,
           256, 128, 256, 128);
      pack(enc_upd_W + ((size_t)enc * 3 + l) * 256 * 128, uW_bf + (size_t)l * 256 * 128,
           256, 128, 256, 128);
    }
    // h0 -> tmpH so that after 3 ping-pong layers the result lands in h_enc
    gemm(xnode_bf, 64, nW_bf, 128, enc_node_b + (size_t)enc * 128,
         nullptr, tmpH, 128, N, 128, 64, FLAG_RELU);
    gemm(xedge_bf, 32, eW_bf, 128, enc_edge_b + (size_t)enc * 128,
         nullptr, e_bf, 128, E, 128, 32, FLAG_RELU);
    bf16_t* cur = tmpH;
    bf16_t* nxt = h_enc;
    for (int l = 0; l < Lnum; ++l) {
      // agg[n] = sum_{e: dst[e]=n} relu([h[src[e]] | e_feat[e]] @ mW + mb)
      fill_f32<<<ew(N * 128), TB, 0, stream>>>(agg, 0.0f, N * 128);
      gemm_seg(cur, src, e_bf, nullptr, nullptr, nullptr, 0,
               mW_bf + (size_t)l * 256 * 128, 128,
               enc_msg_b + ((size_t)enc * 3 + l) * 128,
               agg, nullptr, 128, dstI, E, 128, FLAG_RELU);
      // h' = relu([h | agg] @ uW + ub)   (agg consumed as fp32, converted in-flight)
      gemm_seg(cur, nullptr, (const bf16_t*)agg, nullptr, nullptr, nullptr, 2,
               uW_bf + (size_t)l * 256 * 128, 128,
               enc_upd_b + ((size_t)enc * 3 + l) * 128,
               nullptr, nxt, 128, nullptr, N, 128, FLAG_RELU);
      bf16_t* t2 = cur; cur = nxt; nxt = t2;
    }
  }
  const bf16_t* h_act = hbf;
  const bf16_t* h_del = hbf + (size_t)1 * N * 128;
  const bf16_t* h_add = hbf + (size_t)2 * N * 128;
  const bf16_t* h_arm = hbf + (size_t)3 * N * 128;

  // ---- phase 2: Set2Set readout ----
  fill_f32<<<ew(B * 256), TB, 0, stream>>>(q_star, 0.0f, B * 256);
  fill_f32<<<ew(B * 128), TB, 0, stream>>>(h0, 0.0f, B * 128);
  fill_f32<<<ew(B * 128), TB, 0, stream>>>(c0, 0.0f, B * 128);
  fill_f32<<<ew(B * 128), TB, 0, stream>>>(h1, 0.0f, B * 128);
  fill_f32<<<ew(B * 128), TB, 0, stream>>>(c1, 0.0f, B * 128);
  for (int it = 0; it < 6; ++it) {
    cast(q_star, qs_bf, (int)B, 256, (int)B, 256);
    gemm(qs_bf, 256, Wi0_bf, 512, lstm_b0, zbuf, nullptr, 512, B, 512, 256, 0);
    cast(h0, hB_bf, (int)B, 128, (int)B, 128);
    gemm(hB_bf, 128, Wh0_bf, 512, nullptr, zbuf, nullptr, 512, B, 512, 128, FLAG_LOADC);
    lstm_gate<<<ew(B * 128), TB, 0, stream>>>(zbuf, h0, c0, (int)B);
    cast(h0, hB_bf, (int)B, 128, (int)B, 128);
    gemm(hB_bf, 128, Wi1_bf, 512, lstm_b1, zbuf, nullptr, 512, B, 512, 128, 0);
    cast(h1, hB_bf, (int)B, 128, (int)B, 128);
    gemm(hB_bf, 128, Wh1_bf, 512, nullptr, zbuf, nullptr, 512, B, 512, 128, FLAG_LOADC);
    lstm_gate<<<ew(B * 128), TB, 0, stream>>>(zbuf, h1, c1, (int)B);
    attn_dot<<<ew(N), TB, 0, stream>>>(h_act, h1, gid, ei, N);
    fill_f32<<<ew(B), TB, 0, stream>>>(mB, -3.0e38f, B);
    fill_f32<<<ew(B), TB, 0, stream>>>(sB, 0.0f, B);
    fill_f32<<<ew(B * 128), TB, 0, stream>>>(rB, 0.0f, B * 128);
    seg_max<<<ew(N), TB, 0, stream>>>(ei, gid, mB, N);
    seg_expsum<<<ew(N), TB, 0, stream>>>(ei, gid, mB, exv, sB, N);
    seg_weighted<<<ew(N * 128), TB, 0, stream>>>(h_act, exv, sB, gid, rB, N);
    qstar_update<<<ew(B * 128), TB, 0, stream>>>(q_star, h1, rB, (int)B);
  }

  // ---- phase 3: heads ----
  cast(q_star, qs_bf, (int)B, 256, (int)B, 256);
  gemm(qs_bf, 256, actW1_bf, 256, act_b1, nullptr, actH_bf, 256, B, 256, 256, FLAG_RELU);
  head_dot<<<ew(B * 2), TB, 0, stream>>>(actH_bf, act_W2, act_b2, out_act, B, 256, 2);

  edge_mlp<<<ew(E), TB, 0, stream>>>(x_edge, em_W1, em_b1, em_W2, em_b2, hedge_bf, E);
  // fused 3-segment gather: [h_del[src] | h_edge | h_del[dst]] @ delW1
  gemm_seg(h_del, src, hedge_bf, nullptr, h_del, dstI, 0,
           delW1_bf, 384, del_b1, nullptr, delH, 384, nullptr, E, 384, FLAG_RELU);
  head_dot<<<ew(E), TB, 0, stream>>>(delH, del_W2, del_b2, out_del, E, 384, 1);

  gemm(h_add, 128, addW1_bf, 128, add_b1, nullptr, addH_bf, 128, N, 128, 128, FLAG_RELU);
  head_dot<<<ew(N), TB, 0, stream>>>(addH_bf, add_W2, add_b2, out_add, N, 128, 1);

  gemm(h_arm, 128, armW1_bf, 128, arm_b1, nullptr, armT_bf, 128, N, 128, 128, FLAG_RELU);
  gemm(armT_bf, 128, armW2_bf, 208, arm_b2, out_arm, nullptr, 200, N, 200, 128, 0);
}
